// VGAE_79216376808060
// MI455X (gfx1250) — matrix operand order repaired and checked by reading, not compiled, not run on hardware
//
#include <hip/hip_runtime.h>
#include <hip/hip_bf16.h>
#include <math.h>

typedef __attribute__((ext_vector_type(16))) _Float16 v16h;
typedef __attribute__((ext_vector_type(8)))  float    v8f;

static inline int cdiv(long a, long b) { return (int)((a + b - 1) / b); }

// ---------------- graph-norm kernels ----------------

__global__ void k_zero(float* __restrict__ p, int n) {
  int i = blockIdx.x * blockDim.x + threadIdx.x;
  if (i < n) p[i] = 0.0f;
}

__global__ void k_degcount(const int* __restrict__ ei, float* __restrict__ deg, int E) {
  int e = blockIdx.x * blockDim.x + threadIdx.x;
  if (e < E) atomicAdd(&deg[ei[E + e]], 1.0f);
}

// dinv = (deg + 1)^-1/2   (self-loop makes deg >= 1 always)
__global__ void k_dinv(float* __restrict__ deg, int n) {
  int i = blockIdx.x * blockDim.x + threadIdx.x;
  if (i < n) deg[i] = rsqrtf(deg[i] + 1.0f);
}

__global__ void k_norm(const int* __restrict__ ei, const float* __restrict__ dinv,
                       float* __restrict__ nrm, int E) {
  int e = blockIdx.x * blockDim.x + threadIdx.x;
  if (e < E) nrm[e] = dinv[ei[e]] * dinv[ei[E + e]];
}

// ---------------- weight pre-pack into B-fragment layout ----------------
// Wp[((ct*KB + kb)*32 + lane)*16 + h] = (f16) W[(kb*32 + 8*(lane>>4) + h + (h&8))*cout + ct*16 + (lane&15)]
// so a GEMM wave loads its whole B fragment as one contiguous 32B chunk per lane.

__global__ void k_packW(const float* __restrict__ W, _Float16* __restrict__ Wp,
                        int cin, int cout) {
  int tid = blockIdx.x * blockDim.x + threadIdx.x;
  int KB = cin >> 5;
  int CT = cout >> 4;
  if (tid >= CT * KB * 32) return;
  int lane = tid & 31;
  int idx  = tid >> 5;          // ct*KB + kb
  int kb   = idx % KB;
  int ct   = idx / KB;
  int hs = lane >> 4, lr = lane & 15;
  const float* src = W + (size_t)(kb * 32 + 8 * hs) * cout + ct * 16 + lr;
  _Float16* dst = Wp + (size_t)tid * 16;
#pragma unroll
  for (int h = 0; h < 16; ++h) dst[h] = (_Float16)src[(size_t)(h + (h & 8)) * cout];
}

// ---------------- WMMA GEMM: C[n,COUT] = op(A[n,CIN]) @ W (+bias) ----------------
// One wave per 16-row block. A fragments loaded once into registers; per column
// tile all packed-B fragments are fetched up front (one load clause, staged
// waits) then a fully-unrolled v_wmma_f32_16x16x32_f16 chain accumulates.

template <int CIN, int COUT>
__global__ void k_gemm_wmma(const float* __restrict__ A, const v16h* __restrict__ Wp,
                            const float* __restrict__ bias, float* __restrict__ C,
                            int nrows, int relu_in) {
  constexpr int KB = CIN / 32;    // K blocks of 32
  constexpr int CT = COUT / 16;   // column tiles
  int wid  = (blockIdx.x * blockDim.x + threadIdx.x) >> 5;
  int lane = threadIdx.x & 31;
  int rowBlocks = (nrows + 15) >> 4;
  if (wid >= rowBlocks) return;   // wave-uniform: EXEC stays all-ones below

  int rowBase = wid << 4;
  int hs = lane >> 4;             // 0: lanes 0-15, 1: lanes 16-31
  int lr = lane & 15;
  bool fullTile = (rowBase + 16) <= nrows;   // wave-uniform store guard
  int r = rowBase + lr;
  if (r >= nrows) r = nrows - 1;  // clamp loads
  const float* ar = A + (size_t)r * CIN;

  // A fragments: halves 0-7 -> K = kb*32 + 8*hs + [0..7]; halves 8-15 -> +16
  v16h af[KB];
#pragma unroll
  for (int kb = 0; kb < KB; ++kb) {
    int b0 = kb * 32 + 8 * hs;
    int b1 = b0 + 16;
    float4 p0 = *(const float4*)(ar + b0);
    float4 p1 = *(const float4*)(ar + b0 + 4);
    float4 p2 = *(const float4*)(ar + b1);
    float4 p3 = *(const float4*)(ar + b1 + 4);
    float av[16] = {p0.x, p0.y, p0.z, p0.w, p1.x, p1.y, p1.z, p1.w,
                    p2.x, p2.y, p2.z, p2.w, p3.x, p3.y, p3.z, p3.w};
#pragma unroll
    for (int i = 0; i < 16; ++i) {
      float v = relu_in ? fmaxf(av[i], 0.0f) : av[i];
      af[kb][i] = (_Float16)v;
    }
  }

#pragma unroll
  for (int ct = 0; ct < CT; ++ct) {
    // fetch all B fragments for this tile first (lets loads clause + overlap)
    v16h bf[KB];
#pragma unroll
    for (int kb = 0; kb < KB; ++kb)
      bf[kb] = Wp[(size_t)(ct * KB + kb) * 32 + lane];

    v8f acc = {};
#pragma unroll
    for (int kb = 0; kb < KB; ++kb)
      acc = __builtin_amdgcn_wmma_f32_16x16x32_f16(
          false, af[kb], false, bf[kb], (short)0, acc, false, false);

    float bv = bias ? bias[ct * 16 + lr] : 0.0f;
    float* cp = C + (size_t)(rowBase + 8 * hs) * COUT + ct * 16 + lr;
    if (fullTile) {                 // common path: unguarded, no exec juggling
#pragma unroll
      for (int v = 0; v < 8; ++v) cp[(size_t)v * COUT] = acc[v] + bv;
    } else {
#pragma unroll
      for (int v = 0; v < 8; ++v) {
        int m = rowBase + 8 * hs + v;
        if (m < nrows) C[(size_t)m * COUT + ct * 16 + lr] = acc[v] + bv;
      }
    }
  }
}

// ---------------- conv aggregation ----------------

// out[i,c] = b[c] + dinv[i]^2 * h[i,c]   (bias + self-loop; fully initializes out)
__global__ void k_init_self(float* __restrict__ out, const float* __restrict__ h,
                            const float* __restrict__ b, const float* __restrict__ dinv,
                            int n, int C) {
  int tid = blockIdx.x * blockDim.x + threadIdx.x;
  int qpr = C >> 2;
  if (tid >= n * qpr) return;
  int i = tid / qpr;
  int q = (tid - i * qpr) << 2;
  float di = dinv[i];
  float w = di * di;
  float4 hv = *(const float4*)(h + (size_t)i * C + q);
  float4 bv = *(const float4*)(b + q);
  float4 o;
  o.x = bv.x + w * hv.x;  o.y = bv.y + w * hv.y;
  o.z = bv.z + w * hv.z;  o.w = bv.w + w * hv.w;
  *(float4*)(out + (size_t)i * C + q) = o;
}

// out[dst] += norm[e] * h[src]; thread = (edge, float4 chunk); atomics hit L2.
__global__ void k_scatter(const int* __restrict__ ei, const float* __restrict__ nrm,
                          const float* __restrict__ h, float* __restrict__ out,
                          int E, int C) {
  long tid = (long)blockIdx.x * blockDim.x + threadIdx.x;
  int qpr = C >> 2;
  if (tid >= (long)E * qpr) return;
  int e = (int)(tid / qpr);
  int q = (int)(tid - (long)e * qpr) << 2;
  int s = ei[e];
  int d = ei[E + e];
  float w = nrm[e];
  float4 hv = *(const float4*)(h + (size_t)s * C + q);
  float* op = out + (size_t)d * C + q;
  atomicAdd(op + 0, w * hv.x);
  atomicAdd(op + 1, w * hv.y);
  atomicAdd(op + 2, w * hv.z);
  atomicAdd(op + 3, w * hv.w);
}

// z = mu + eps * exp(0.5 * logvar)
__global__ void k_reparam(const float* __restrict__ mu, const float* __restrict__ lv,
                          const float* __restrict__ eps, float* __restrict__ z, int total) {
  int i = blockIdx.x * blockDim.x + threadIdx.x;
  if (i < total) z[i] = mu[i] + eps[i] * expf(0.5f * lv[i]);
}

// ---------------- host-side launch ----------------

extern "C" void kernel_launch(void* const* d_in, const int* in_sizes, int n_in,
                              void* d_out, int out_size, void* d_ws, size_t ws_size,
                              hipStream_t stream) {
  const float* x    = (const float*)d_in[0];
  const int*   ei   = (const int*)d_in[1];
  const float* eps  = (const float*)d_in[2];
  const float* W_e1 = (const float*)d_in[3];
  const float* b_e1 = (const float*)d_in[4];
  const float* W_e2 = (const float*)d_in[5];
  const float* b_e2 = (const float*)d_in[6];
  const float* W_mu = (const float*)d_in[7];
  const float* b_mu = (const float*)d_in[8];
  const float* W_lv = (const float*)d_in[9];
  const float* b_lv = (const float*)d_in[10];
  const float* W_d1 = (const float*)d_in[11];
  const float* b_d1 = (const float*)d_in[12];
  const float* W_d2 = (const float*)d_in[13];
  const float* b_d2 = (const float*)d_in[14];

  const int IN_C = 128, OUT_C = 64;
  const int N = in_sizes[0] / IN_C;
  const int E = in_sizes[1] / 2;

  // workspace carve-out (256B-aligned)
  char* wsb = (char*)d_ws;
  size_t off = 0;
  auto take = [&](size_t bytes) { size_t o = off; off = (off + bytes + 255) & ~(size_t)255; return o; };
  float*    dinv  = (float*)(wsb + take((size_t)N * 4));
  float*    nrm   = (float*)(wsb + take((size_t)E * 4));
  float*    gA    = (float*)(wsb + take((size_t)N * IN_C * 4));
  float*    gB    = (float*)(wsb + take((size_t)N * IN_C * 4));
  _Float16* wpE1  = (_Float16*)(wsb + take((size_t)IN_C * IN_C * 2));
  _Float16* wpE2  = (_Float16*)(wsb + take((size_t)IN_C * OUT_C * 2));
  _Float16* wpMu  = (_Float16*)(wsb + take((size_t)OUT_C * OUT_C * 2));
  _Float16* wpLv  = (_Float16*)(wsb + take((size_t)OUT_C * OUT_C * 2));
  _Float16* wpD1  = (_Float16*)(wsb + take((size_t)OUT_C * IN_C * 2));
  _Float16* wpD2  = (_Float16*)(wsb + take((size_t)IN_C * IN_C * 2));

  float* outD  = (float*)d_out;                       // [N,128]
  float* outMu = outD + (size_t)N * IN_C;             // [N,64]
  float* outLv = outMu + (size_t)N * OUT_C;           // [N,64]

  const int T = 256;
  auto pack = [&](const float* W, _Float16* Wp, int cin, int cout) {
    int total = (cout / 16) * (cin / 32) * 32;
    k_packW<<<cdiv(total, T), T, 0, stream>>>(W, Wp, cin, cout);
  };
  long gw = ((N + 15) / 16) * 32;  // one wave per 16-row block
  auto initself = [&](float* out, const float* h, const float* b, int C) {
    k_init_self<<<cdiv((long)N * (C / 4), T), T, 0, stream>>>(out, h, b, dinv, N, C);
  };
  auto scatter = [&](const float* h, float* out, int C) {
    k_scatter<<<cdiv((long)E * (C / 4), T), T, 0, stream>>>(ei, nrm, h, out, E, C);
  };

  // pack all weights (tiny, deterministic each call)
  pack(W_e1, wpE1, IN_C, IN_C);
  pack(W_e2, wpE2, IN_C, OUT_C);
  pack(W_mu, wpMu, OUT_C, OUT_C);
  pack(W_lv, wpLv, OUT_C, OUT_C);
  pack(W_d1, wpD1, OUT_C, IN_C);
  pack(W_d2, wpD2, IN_C, IN_C);

  // gcn_norm
  k_zero<<<cdiv(N, T), T, 0, stream>>>(dinv, N);
  k_degcount<<<cdiv(E, T), T, 0, stream>>>(ei, dinv, E);
  k_dinv<<<cdiv(N, T), T, 0, stream>>>(dinv, N);
  k_norm<<<cdiv(E, T), T, 0, stream>>>(ei, dinv, nrm, E);

  // encoder conv1: h1 = relu(agg(x @ W_e1) + b_e1)   (relu deferred into next GEMM)
  k_gemm_wmma<128, 128><<<cdiv(gw, T), T, 0, stream>>>(x, (const v16h*)wpE1, nullptr, gA, N, 0);
  initself(gB, gA, b_e1, IN_C);
  scatter(gA, gB, IN_C);

  // encoder conv2: h2 = agg(relu(h1) @ W_e2) + b_e2
  k_gemm_wmma<128, 64><<<cdiv(gw, T), T, 0, stream>>>(gB, (const v16h*)wpE2, nullptr, gA, N, 1);
  initself(gB, gA, b_e2, OUT_C);
  scatter(gA, gB, OUT_C);                              // gB[:N*64] = h2

  // mu / logvar straight into d_out slices (bias fused)
  k_gemm_wmma<64, 64><<<cdiv(gw, T), T, 0, stream>>>(gB, (const v16h*)wpMu, b_mu, outMu, N, 0);
  k_gemm_wmma<64, 64><<<cdiv(gw, T), T, 0, stream>>>(gB, (const v16h*)wpLv, b_lv, outLv, N, 0);

  // z = mu + eps * exp(0.5*logvar)  -> gA[:N*64]
  k_reparam<<<cdiv((long)N * OUT_C, T), T, 0, stream>>>(outMu, outLv, eps, gA, N * OUT_C);

  // decoder conv1: d1 = relu(agg(z @ W_d1) + b_d1)   (relu deferred)
  k_gemm_wmma<64, 128><<<cdiv(gw, T), T, 0, stream>>>(gA, (const v16h*)wpD1, nullptr, gB, N, 0);
  initself(gA, gB, b_d1, IN_C);
  scatter(gB, gA, IN_C);

  // decoder conv2: d = agg(relu(d1) @ W_d2) + b_d2  -> d_out
  k_gemm_wmma<128, 128><<<cdiv(gw, T), T, 0, stream>>>(gA, (const v16h*)wpD2, nullptr, gB, N, 1);
  initself(outD, gB, b_d2, IN_C);
  scatter(gB, outD, IN_C);
}
